// CorrelationLayer_90074054132600
// MI455X (gfx1250) — compile-verified, matrix operand order checked by reading, not run on hardware
//
#include <hip/hip_runtime.h>
#include <math.h>

typedef __bf16 bf16_t;
typedef __attribute__((ext_vector_type(16))) __bf16 v16bf;
typedef __attribute__((ext_vector_type(8)))  float  v8f;

#define B_    8
#define C_    256
#define H_    96
#define W_    96
#define PADP  20
#define HP    136
#define WP    136
#define NOFF  21      // displacements 0..40 step 2
#define ND    441
#define ASTRIDE 264   // 256 + 8 bf16 pad -> conflict-free ds_load_b128
#define BSTRIDE 264

// ---- workspace layout (bytes) ----
// sub1 : B*H*W*C  bf16 = 37,748,736
// sub2 : B*HP*WP*C bf16 = 75,759,616
// std1 : B*H*W   f32    =    294,912
// std2 : B*HP*WP f32    =    591,872   (total ~114.4 MB)
#define OFF_SUB2 37748736ULL
#define OFF_STD1 (OFF_SUB2 + 75759616ULL)
#define OFF_STD2 (OFF_STD1 + 294912ULL)

// ---------------- staging: x1 -> channels-last bf16 (mean-sub) + std ----------------
__global__ __launch_bounds__(256) void stage_x1(const float* __restrict__ x1,
                                                bf16_t* __restrict__ sub1,
                                                float* __restrict__ std1) {
  int idx = blockIdx.x * 256 + threadIdx.x;        // pixel = b*9216 + (h*96+w)
  if (idx >= B_ * H_ * W_) return;
  int b = idx / (H_ * W_);
  int s = idx - b * (H_ * W_);
  const float* src = x1 + (size_t)b * C_ * H_ * W_ + s;
  float sum = 0.f, sq = 0.f;
  for (int c = 0; c < C_; ++c) {                   // coalesced over consecutive pixels
    float v = src[(size_t)c * (H_ * W_)];
    sum += v; sq += v * v;
  }
  float mean = sum * (1.0f / C_);
  float var  = sq - (float)C_ * mean * mean;
  if (var < 0.f) var = 0.f;
  std1[idx] = sqrtf(var * (1.0f / (C_ - 1)));
  bf16_t* dst = sub1 + (size_t)idx * C_;
  for (int c = 0; c < C_; ++c) {
    float v = src[(size_t)c * (H_ * W_)];
    dst[c] = (bf16_t)(v - mean);
  }
}

// ---------------- staging: x2 -> zero-padded channels-last bf16 + std ----------------
__global__ __launch_bounds__(256) void stage_x2(const float* __restrict__ x2,
                                                bf16_t* __restrict__ sub2,
                                                float* __restrict__ std2) {
  int idx = blockIdx.x * 256 + threadIdx.x;        // padded pixel = b*18496 + (hp*136+wp)
  if (idx >= B_ * HP * WP) return;
  int b  = idx / (HP * WP);
  int s2 = idx - b * (HP * WP);
  int hp = s2 / WP, wp = s2 - hp * WP;
  int h = hp - PADP, w = wp - PADP;
  bf16_t* dst = sub2 + (size_t)idx * C_;
  if ((unsigned)h >= (unsigned)H_ || (unsigned)w >= (unsigned)W_) {
    std2[idx] = 0.f;                               // zero pad: sub=0, std=0 (0/0 -> NaN like ref)
    for (int c = 0; c < C_; ++c) dst[c] = (bf16_t)0.f;
    return;
  }
  const float* src = x2 + (size_t)b * C_ * H_ * W_ + h * W_ + w;
  float sum = 0.f, sq = 0.f;
  for (int c = 0; c < C_; ++c) {
    float v = src[(size_t)c * (H_ * W_)];
    sum += v; sq += v * v;
  }
  float mean = sum * (1.0f / C_);
  float var  = sq - (float)C_ * mean * mean;
  if (var < 0.f) var = 0.f;
  std2[idx] = sqrtf(var * (1.0f / (C_ - 1)));
  for (int c = 0; c < C_; ++c) {
    float v = src[(size_t)c * (H_ * W_)];
    dst[c] = (bf16_t)(v - mean);
  }
}

// ---------------- main: Gram-tile WMMA + stride-2 diagonal extraction ----------------
// block = 128 threads (4 waves). blockIdx = (w-tile, h, b).
// For each of 21 row-offsets ox: G[16 x 64] = A(16x256) * B(64x256)^T via
// v_wmma_f32_16x16x32_bf16; scatter entries with (n-m) even in [0,40].
__global__ __launch_bounds__(128) void corr_wmma(const bf16_t* __restrict__ sub1,
                                                 const bf16_t* __restrict__ sub2,
                                                 const float* __restrict__ std1,
                                                 const float* __restrict__ std2,
                                                 float* __restrict__ out) {
  const int w0   = blockIdx.x << 4;
  const int h    = blockIdx.y;
  const int b    = blockIdx.z;
  const int t    = threadIdx.x;
  const int lane = t & 31;
  const int wave = t >> 5;

  __shared__ bf16_t Ash[16 * ASTRIDE];
  __shared__ bf16_t Bsh[64 * BSTRIDE];

  // --- load A tile: 16 consecutive pixels x 256 ch (contiguous, 512 x uint4) ---
  {
    const uint4* Ag = reinterpret_cast<const uint4*>(
        sub1 + ((size_t)(b * (H_ * W_) + h * W_ + w0)) * C_);
    for (int i = t; i < 512; i += 128) {
      int e = i << 3;                 // bf16 element index in packed 16x256
      int m = e >> 8, c = e & 255;
      *reinterpret_cast<uint4*>(&Ash[m * ASTRIDE + c]) = Ag[i];
    }
  }

  const int mrow = lane & 15;             // A row / C-tile column-within-lane group
  const int hi   = lane >> 4;             // lane half selects K/M sub-block
  const int nl   = (wave << 4) | mrow;    // this lane's N column (0..63)

  for (int oxi = 0; oxi < NOFF; ++oxi) {
    const int h2 = h + (oxi << 1);        // padded row, <= 95+40=135
    __syncthreads();                      // prev iter's LDS reads done
    {
      // --- load B tile: 64 pixels x 256 ch of padded row h2, zero-fill past WP ---
      const uint4* Bg = reinterpret_cast<const uint4*>(
          sub2 + ((size_t)((b * HP + h2) * WP + w0)) * C_);
      int npix = WP - w0; if (npix > 64) npix = 64;
      int nvec = npix * (C_ / 8);
      for (int i = t; i < 2048; i += 128) {
        uint4 v;
        if (i < nvec) v = Bg[i];
        else { v.x = 0u; v.y = 0u; v.z = 0u; v.w = 0u; }
        int e = i << 3;
        int n = e >> 8, c = e & 255;
        *reinterpret_cast<uint4*>(&Bsh[n * BSTRIDE + c]) = v;
      }
    }
    __syncthreads();

    v8f acc = {0.f, 0.f, 0.f, 0.f, 0.f, 0.f, 0.f, 0.f};
#pragma unroll
    for (int c0 = 0; c0 < C_; c0 += 32) {
      // 16-bit A layout: lane(0..15)=row M, elems 0..7 -> K = 8*hi+0..7,
      // elems 8..15 -> K = 8*hi+16..23 ; B mirrors with lane = column N.
      v16bf af, bfr;
      const bf16_t* ap = &Ash[mrow * ASTRIDE + c0 + (hi << 3)];
      *reinterpret_cast<uint4*>(&af)       = *reinterpret_cast<const uint4*>(ap);
      *(reinterpret_cast<uint4*>(&af) + 1) = *reinterpret_cast<const uint4*>(ap + 16);
      const bf16_t* bp = &Bsh[nl * BSTRIDE + c0 + (hi << 3)];
      *reinterpret_cast<uint4*>(&bfr)       = *reinterpret_cast<const uint4*>(bp);
      *(reinterpret_cast<uint4*>(&bfr) + 1) = *reinterpret_cast<const uint4*>(bp + 16);
      acc = __builtin_amdgcn_wmma_f32_16x16x32_bf16(false, af, false, bfr,
                                                    (short)0, acc, false, false);
    }

    // --- scatter diagonals: lane holds D[M = v+8*hi, N = nl] ---
    const int w2c = w0 + nl;
    float s2v = std2[(size_t)(b * HP + h2) * WP + (w2c < WP ? w2c : WP - 1)];
#pragma unroll
    for (int v = 0; v < 8; ++v) {
      int m    = v + (hi << 3);
      int diff = nl - m;                                 // = oy
      if (diff >= 0 && diff <= 40 && ((diff & 1) == 0)) {
        int w = w0 + m;
        float s1v = std1[(size_t)b * (H_ * W_) + h * W_ + w];
        int d = (diff >> 1) * NOFF + oxi;                // grid order: oy outer, ox inner
        out[(((size_t)b * ND + d) * H_ + h) * W_ + w] = acc[v] / (s1v * s2v);
      }
    }
  }
}

extern "C" void kernel_launch(void* const* d_in, const int* in_sizes, int n_in,
                              void* d_out, int out_size, void* d_ws, size_t ws_size,
                              hipStream_t stream) {
  (void)in_sizes; (void)n_in; (void)out_size; (void)ws_size;
  const float* x1 = (const float*)d_in[0];
  const float* x2 = (const float*)d_in[1];
  float* out = (float*)d_out;
  char* ws = (char*)d_ws;
  bf16_t* sub1 = (bf16_t*)ws;
  bf16_t* sub2 = (bf16_t*)(ws + OFF_SUB2);
  float*  std1 = (float*)(ws + OFF_STD1);
  float*  std2 = (float*)(ws + OFF_STD2);

  stage_x1<<<(B_ * H_ * W_ + 255) / 256, 256, 0, stream>>>(x1, sub1, std1);
  stage_x2<<<(B_ * HP * WP + 255) / 256, 256, 0, stream>>>(x2, sub2, std2);

  dim3 grid(W_ / 16, H_, B_);   // 6 x 96 x 8 = 4608 workgroups, 4 waves each
  corr_wmma<<<grid, 128, 0, stream>>>(sub1, sub2, std1, std2, out);
}
